// Physics_Attention_Structured_Mesh_2D_61649960567201
// MI455X (gfx1250) — compile-verified, matrix operand order checked by reading, not run on hardware
//
#include <hip/hip_runtime.h>

typedef __attribute__((ext_vector_type(16))) __bf16 v16bf;
typedef __attribute__((ext_vector_type(8)))  __bf16 v8bf;
typedef __attribute__((ext_vector_type(8)))  float  v8f;
typedef __attribute__((ext_vector_type(8)))  unsigned short v8us;
typedef __attribute__((ext_vector_type(4)))  int v4i;

#define HEADS 8
#define DH 64
#define DIMC 128
#define INNER 512
#define BB 4
#define NPIX 16384   // 128*128

#if __has_builtin(__builtin_amdgcn_global_load_async_to_lds_b128) && __has_builtin(__builtin_amdgcn_s_wait_asynccnt)
#define HAVE_ASYNC_LDS 1
#endif

#define AS_GLOBAL __attribute__((address_space(1)))
#define AS_LDS    __attribute__((address_space(3)))

__device__ __forceinline__ unsigned short f2bf(float f) {
    unsigned u = __float_as_uint(f);
    u += 0x7FFFu + ((u >> 16) & 1u);
    return (unsigned short)(u >> 16);
}

__device__ __forceinline__ v16bf mk16(v8us lo, v8us hi) {
    union { v8us u; v8bf b; } a, c;
    a.u = lo; c.u = hi;
    return __builtin_shufflevector(a.b, c.b, 0,1,2,3,4,5,6,7,8,9,10,11,12,13,14,15);
}
// A/B fragment, 16-bit: elements 0..7 at p, elements 8..15 at p+16 (k = kb+e / 16+kb+e)
__device__ __forceinline__ v16bf ldfrag_split(const unsigned short* p) {
    return mk16(*(const v8us*)p, *(const v8us*)(p + 16));
}
// pre-swizzled fragment: 16 contiguous bf16 per lane
__device__ __forceinline__ v16bf ldfrag16(const unsigned short* p) {
    return mk16(*(const v8us*)p, *(const v8us*)(p + 8));
}
#define WMMA_BF16(A,B,C) __builtin_amdgcn_wmma_f32_16x16x32_bf16(false,(A),false,(B),(short)0,(C),false,false)
#define RFL(x) __builtin_amdgcn_readfirstlane(x)

// ---------------- prep kernels ----------------
__global__ void k_cvt_x(const float* __restrict__ x, unsigned short* __restrict__ xb, int n) {
    int i = blockIdx.x * 256 + threadIdx.x;
    if (i < n) xb[i] = f2bf(x[i]);
}

// conv weights -> B-frag layout: [f(64 ofrags)][tap(9)][kc(4)][lane(32)][e(16)]
__global__ void k_prep_wconv(const float* __restrict__ Wx, const float* __restrict__ Wfx,
                             unsigned short* __restrict__ wcat) {
    int i = blockIdx.x * 256 + threadIdx.x;
    if (i >= 64 * 9 * 4 * 32 * 16) return;
    int e = i & 15; int t = i >> 4;
    int lane = t & 31; t >>= 5;
    int kc = t & 3; t >>= 2;
    int tap = t % 9; int f = t / 9;
    int lh = lane >> 4;
    int k = (e < 8) ? (lh * 8 + e) : (16 + lh * 8 + (e - 8));
    int ci = kc * 32 + k;
    int o = f * 16 + (lane & 15);
    int ky = tap / 3, kx = tap % 3;
    float v = (o < 512) ? Wx[((o * DIMC + ci) * 3 + ky) * 3 + kx]
                        : Wfx[(((o - 512) * DIMC + ci) * 3 + ky) * 3 + kx];
    wcat[i] = f2bf(v);
}

// Ws -> B-frag layout: [gf(4)][kc(2)][lane(32)][e(16)]
__global__ void k_prep_ws(const float* __restrict__ Ws, unsigned short* __restrict__ wsb) {
    int i = blockIdx.x * 256 + threadIdx.x;
    if (i >= 4 * 2 * 32 * 16) return;
    int e = i & 15; int t = i >> 4;
    int lane = t & 31; t >>= 5;
    int kc = t & 1; int gf = t >> 1;
    int lh = lane >> 4;
    int k = (e < 8) ? (lh * 8 + e) : (16 + lh * 8 + (e - 8));
    int c = kc * 32 + k;
    int g = gf * 16 + (lane & 15);
    wsb[i] = f2bf(Ws[g * 64 + c]);
}

__global__ void k_zero(float* __restrict__ p, int n) {
    int i = blockIdx.x * 256 + threadIdx.x;
    if (i < n) p[i] = 0.0f;
}

// ---------------- implicit-GEMM conv (both convs fused: 1024 out channels) ----------------
// grid.x = ((b*128)+yy)*2 + half  (1024), grid.y = ot (8 col-tiles of 128)
__global__ __launch_bounds__(256)
void k_conv(const unsigned short* __restrict__ xb,
            const unsigned short* __restrict__ wcat,
            const float* __restrict__ bx, const float* __restrict__ bfx,
            unsigned short* __restrict__ xmid, unsigned short* __restrict__ fxmid) {
    __shared__ unsigned short lin[3 * 66 * 128];   // 3 haloed input rows, bf16
    const int tid = threadIdx.x;
    const int bidx = blockIdx.x;
    const int ot = blockIdx.y;
    const int half = bidx & 1;
    const int yy = (bidx >> 1) & 127;
    const int b = bidx >> 8;
    const int xbase = half * 64;

#ifdef HAVE_ASYNC_LDS
    // async global->LDS staging, 16B per lane; halo/OOB slots pre-zeroed with plain stores
    AS_LDS unsigned short* lin3 = (AS_LDS unsigned short*)lin;
    for (int idx = tid; idx < 3 * 66 * 16; idx += 256) {
        int ky = idx / (66 * 16);
        int rem = idx - ky * 66 * 16;
        int xl = rem >> 4;
        int ch = (rem & 15) * 8;
        int gx = xbase + xl - 1;
        int gy = yy + ky - 1;
        int loff = (ky * 66 + xl) * 128 + ch;
        if (gx >= 0 && gx < 128 && gy >= 0 && gy < 128) {
            const unsigned short* gp = xb + (((b * 128 + gy) * 128 + gx) << 7) + ch;
            __builtin_amdgcn_global_load_async_to_lds_b128(
                (AS_GLOBAL v4i*)(unsigned long long)gp,
                (AS_LDS v4i*)(lin3 + loff), 0, 0);
        } else {
            v8us z = {};
            *(v8us*)&lin[loff] = z;
        }
    }
    __builtin_amdgcn_s_wait_asynccnt(0);
#else
    for (int idx = tid; idx < 3 * 66 * 128; idx += 256) {
        int ky = idx / (66 * 128);
        int rem = idx - ky * 66 * 128;
        int xl = rem >> 7;
        int c = rem & 127;
        int gx = xbase + xl - 1;
        int gy = yy + ky - 1;
        unsigned short v = 0;
        if (gx >= 0 && gx < 128 && gy >= 0 && gy < 128)
            v = xb[(((b * 128 + gy) * 128 + gx) << 7) + c];
        lin[idx] = v;
    }
#endif
    __syncthreads();

    const int wv = tid >> 5, ln = tid & 31;
    const int l16 = ln & 15, lh = ln >> 4, kb = lh * 8;
    const int mbs = RFL((wv >> 1) * 16);           // row group base (wave-uniform -> SGPR)
    const int fgb = RFL(ot * 8 + (wv & 1) * 4);    // ofrag base (wave-uniform -> SGPR)
    const unsigned short* wlane = wcat + ln * 16;  // per-lane constant offset

    v8f acc[4] = {};

    for (int tap = 0; tap < 9; ++tap) {
        const int ky = tap / 3, kx = tap - ky * 3;
        const unsigned short* arow = &lin[(ky * 66 + mbs + l16 + kx) * 128 + kb];
        #pragma unroll
        for (int kc = 0; kc < 4; ++kc) {
            v16bf af = ldfrag_split(arow + kc * 32);
            #pragma unroll
            for (int nf = 0; nf < 4; ++nf) {
                v16bf bfrag = ldfrag16(wlane + (size_t)(((fgb + nf) * 9 + tap) * 4 + kc) * 512);
                acc[nf] = WMMA_BF16(af, bfrag, acc[nf]);
            }
        }
    }

    const int cg = wv & 1;
    #pragma unroll
    for (int nf = 0; nf < 4; ++nf) {
        int o = ot * 128 + cg * 64 + nf * 16 + l16;
        float bias = (o < 512) ? bx[o] : bfx[o - 512];
        unsigned short* dst = (o < 512) ? xmid : fxmid;
        int oc = o & 511;
        int head = oc >> 6, c = oc & 63;
        #pragma unroll
        for (int r = 0; r < 8; ++r) {
            int xpix = xbase + mbs + r + 8 * lh;
            int n = yy * 128 + xpix;
            dst[(((size_t)(b * 8 + head) * NPIX + n) << 6) + c] = f2bf(acc[nf][r] + bias);
        }
    }
}

// ---------------- routing: logits WMMA + register softmax + token-gather WMMA ----------------
// grid.x = ntile (128), grid.y = b*8+h (32)
__global__ __launch_bounds__(256)
void k_route(const unsigned short* __restrict__ xmid,
             const unsigned short* __restrict__ fxmid,
             const unsigned short* __restrict__ wsb,
             const float* __restrict__ bs, const float* __restrict__ temperature,
             unsigned short* __restrict__ sw,
             float* __restrict__ tokacc, float* __restrict__ nrm) {
    __shared__ unsigned short wT[64 * 128];    // [g][n]
    __shared__ unsigned short fxT[64 * 128];   // [c][n]
    const int tid = threadIdx.x;
    const int bh = blockIdx.y;
    const int b = bh >> 3, h = bh & 7;
    const int n0 = blockIdx.x * 128;
    const int wv = tid >> 5, ln = tid & 31, l16 = ln & 15, lh = ln >> 4, kb = lh * 8;

    {   // fx tile -> transposed LDS
        int row = tid >> 1;
        int cb = (tid & 1) * 32;
        const unsigned short* src = fxmid + ((size_t)bh * NPIX + n0 + row) * 64 + cb;
        #pragma unroll
        for (int i = 0; i < 4; ++i) {
            v8us v = *(const v8us*)(src + i * 8);
            #pragma unroll
            for (int j = 0; j < 8; ++j)
                fxT[(cb + i * 8 + j) * 128 + row] = v[j];
        }
    }

    // logits: each wave 16 rows x 64 g
    const int nbs = RFL(wv * 16);
    v8f acc[4] = {};
    {
        const unsigned short* abase = xmid + ((size_t)bh * NPIX + n0 + nbs) * 64;  // scalar
        const int aoff = l16 * 64 + kb;                                            // vector, invariant
        const unsigned short* wlane = wsb + ln * 16;
        #pragma unroll
        for (int kc = 0; kc < 2; ++kc) {
            v16bf af = ldfrag_split(abase + aoff + kc * 32);
            #pragma unroll
            for (int gf = 0; gf < 4; ++gf) {
                v16bf bfrag = ldfrag16(wlane + (gf * 2 + kc) * 512);
                acc[gf] = WMMA_BF16(af, bfrag, acc[gf]);
            }
        }
    }

    float tmp = temperature[h];
    tmp = fminf(fmaxf(tmp, 0.1f), 5.0f);
    float invt = 1.0f / tmp;

    float val[4][8];
    #pragma unroll
    for (int gf = 0; gf < 4; ++gf) {
        float bsv = bs[gf * 16 + l16];
        #pragma unroll
        for (int r = 0; r < 8; ++r) val[gf][r] = (acc[gf][r] + bsv) * invt;
    }

    // softmax over g=64 (4 frags x 16 lanes within the lane-half)
    float nsum[4] = {0.f, 0.f, 0.f, 0.f};
    #pragma unroll
    for (int r = 0; r < 8; ++r) {
        float m = val[0][r];
        #pragma unroll
        for (int gf = 1; gf < 4; ++gf) m = fmaxf(m, val[gf][r]);
        for (int msk = 8; msk >= 1; msk >>= 1) m = fmaxf(m, __shfl_xor(m, msk, 32));
        float s = 0.f;
        #pragma unroll
        for (int gf = 0; gf < 4; ++gf) { val[gf][r] = __expf(val[gf][r] - m); s += val[gf][r]; }
        for (int msk = 8; msk >= 1; msk >>= 1) s += __shfl_xor(s, msk, 32);
        float inv = 1.0f / s;
        #pragma unroll
        for (int gf = 0; gf < 4; ++gf) {
            float w = val[gf][r] * inv;
            val[gf][r] = w;
            nsum[gf] += w;
        }
    }

    #pragma unroll
    for (int gf = 0; gf < 4; ++gf) {
        int g = gf * 16 + l16;
        #pragma unroll
        for (int r = 0; r < 8; ++r) {
            int nl = nbs + r + 8 * lh;
            unsigned short wbits = f2bf(val[gf][r]);
            wT[g * 128 + nl] = wbits;
            sw[((size_t)b * NPIX + n0 + nl) * 512 + h * 64 + g] = wbits;
        }
        atomicAdd(&nrm[bh * 64 + g], nsum[gf]);
    }
    __syncthreads();

    // token partial: (64g x 64c) += wT(64x128) x fxT^T(128x64), per wave 16g x 32c
    const int gbs = RFL((wv >> 1) * 16);
    const int cb2 = RFL((wv & 1) * 32);
    v8f tac[2] = {};
    #pragma unroll
    for (int nc = 0; nc < 4; ++nc) {
        v16bf af = ldfrag_split(&wT[(gbs + l16) * 128 + nc * 32 + kb]);
        #pragma unroll
        for (int fb = 0; fb < 2; ++fb) {
            v16bf bfrag = ldfrag_split(&fxT[(cb2 + fb * 16 + l16) * 128 + nc * 32 + kb]);
            tac[fb] = WMMA_BF16(af, bfrag, tac[fb]);
        }
    }
    #pragma unroll
    for (int fb = 0; fb < 2; ++fb) {
        int c = cb2 + fb * 16 + l16;
        #pragma unroll
        for (int r = 0; r < 8; ++r) {
            int g = gbs + r + 8 * lh;
            atomicAdd(&tokacc[((size_t)bh * 64 + g) * 64 + c], tac[fb][r]);
        }
    }
}

// ---------------- tiny attention + P precompute (negligible FLOPs) ----------------
// grid = 32 (b*8+h), 64 threads (one per slice token g)
__global__ __launch_bounds__(64)
void k_attn(const float* __restrict__ tokacc, const float* __restrict__ nrm,
            const float* __restrict__ Wq, const float* __restrict__ Wk,
            const float* __restrict__ Wv, const float* __restrict__ Wo,
            unsigned short* __restrict__ PB) {
    __shared__ float tok[64 * 64];
    __shared__ float kk[64 * 64];
    __shared__ float vv[64 * 64];
    const int bh = blockIdx.x; const int b = bh >> 3, h = bh & 7;
    const int g = threadIdx.x;
    float inv = 1.0f / (nrm[bh * 64 + g] + 1e-5f);
    for (int c = 0; c < 64; ++c)
        tok[g * 64 + c] = tokacc[((size_t)bh * 64 + g) * 64 + c] * inv;
    __syncthreads();
    float qr[64];
    for (int d = 0; d < 64; ++d) {
        float q = 0.f, k = 0.f, v = 0.f;
        for (int c = 0; c < 64; ++c) {
            float t = tok[g * 64 + c];
            q += t * Wq[d * 64 + c]; k += t * Wk[d * 64 + c]; v += t * Wv[d * 64 + c];
        }
        qr[d] = q; kk[g * 64 + d] = k; vv[g * 64 + d] = v;
    }
    __syncthreads();
    float att[64]; float mx = -3.0e38f;
    for (int j = 0; j < 64; ++j) {
        float s = 0.f;
        for (int d = 0; d < 64; ++d) s += qr[d] * kk[j * 64 + d];
        s *= 0.125f;
        att[j] = s; mx = fmaxf(mx, s);
    }
    float ssum = 0.f;
    for (int j = 0; j < 64; ++j) { att[j] = __expf(att[j] - mx); ssum += att[j]; }
    float isum = 1.0f / ssum;
    float outr[64];
    for (int d = 0; d < 64; ++d) {
        float o = 0.f;
        for (int j = 0; j < 64; ++j) o += att[j] * vv[j * 64 + d];
        outr[d] = o * isum;
    }
    // P[g][o] = sum_c out[g][c]*Wo[o, h*64+c], stored pre-swizzled for B-fragments
    int jj = h * 64 + g; int kc = jj >> 5; int k = jj & 31;
    int lh = (k >> 3) & 1; int e = (k < 16) ? (k & 7) : (8 + (k & 7));
    for (int o = 0; o < 128; ++o) {
        float p = 0.f;
        for (int c = 0; c < 64; ++c) p += outr[c] * Wo[o * 512 + h * 64 + c];
        int lane = (o & 15) | (lh << 4); int of = o >> 4;
        PB[(((size_t)(b * 8 + of) * 16 + kc) * 512) + lane * 16 + e] = f2bf(p);
    }
}

// ---------------- fused scatter + output projection: out = sw(Nx512) x P(512x128) + bo ----------------
// grid = 512 (b * 128 row-tiles)
__global__ __launch_bounds__(256)
void k_outproj(const unsigned short* __restrict__ sw,
               const unsigned short* __restrict__ PB,
               const float* __restrict__ bo,
               float* __restrict__ out) {
    const int tid = threadIdx.x;
    const int blk = blockIdx.x;
    const int b = blk >> 7;
    const int n0 = (blk & 127) * 128;
    const int wv = tid >> 5, ln = tid & 31, l16 = ln & 15, lh = ln >> 4, kb = lh * 8;
    const int mbs = RFL((wv >> 1) * 32);
    const int cgs = RFL(wv & 1);

    const unsigned short* abase = sw + ((size_t)b * NPIX + n0 + mbs) * 512;  // scalar
    const int aoff = l16 * 512 + kb;                                         // vector, invariant
    const unsigned short* pblane = PB + ln * 16;                             // vector, invariant

    v8f acc[2][4] = {};
    for (int kc = 0; kc < 16; ++kc) {
        v16bf af[2];
        #pragma unroll
        for (int mi = 0; mi < 2; ++mi)
            af[mi] = ldfrag_split(abase + mi * (16 * 512) + kc * 32 + aoff);
        #pragma unroll
        for (int nf = 0; nf < 4; ++nf) {
            v16bf bfrag = ldfrag16(pblane + ((size_t)(b * 8 + cgs * 4 + nf) * 16 + kc) * 512);
            #pragma unroll
            for (int mi = 0; mi < 2; ++mi)
                acc[mi][nf] = WMMA_BF16(af[mi], bfrag, acc[mi][nf]);
        }
    }
    #pragma unroll
    for (int nf = 0; nf < 4; ++nf) {
        int o = cgs * 64 + nf * 16 + l16;
        float bias = bo[o];
        #pragma unroll
        for (int mi = 0; mi < 2; ++mi)
            #pragma unroll
            for (int r = 0; r < 8; ++r) {
                int n = n0 + mbs + mi * 16 + r + 8 * lh;
                out[((size_t)b * NPIX + n) * 128 + o] = acc[mi][nf][r] + bias;
            }
    }
}

// ---------------- host ----------------
extern "C" void kernel_launch(void* const* d_in, const int* in_sizes, int n_in,
                              void* d_out, int out_size, void* d_ws, size_t ws_size,
                              hipStream_t stream) {
    const float* x    = (const float*)d_in[0];
    const float* temperature = (const float*)d_in[1];
    const float* Wx   = (const float*)d_in[2];
    const float* bx   = (const float*)d_in[3];
    const float* Wfx  = (const float*)d_in[4];
    const float* bfx  = (const float*)d_in[5];
    const float* Ws   = (const float*)d_in[6];
    const float* bs   = (const float*)d_in[7];
    const float* Wq   = (const float*)d_in[8];
    const float* Wk   = (const float*)d_in[9];
    const float* Wv   = (const float*)d_in[10];
    const float* Wo   = (const float*)d_in[11];
    const float* bo   = (const float*)d_in[12];
    float* out = (float*)d_out;

    char* wsb_base = (char*)d_ws;
    size_t off = 0;
    auto alloc = [&](size_t bytes) -> char* {
        char* p = wsb_base + off;
        off = (off + bytes + 255) & ~(size_t)255;
        return p;
    };
    unsigned short* xb    = (unsigned short*)alloc((size_t)BB * NPIX * DIMC * 2);
    unsigned short* wcat  = (unsigned short*)alloc((size_t)64 * 9 * 4 * 32 * 16 * 2);
    unsigned short* wsbuf = (unsigned short*)alloc((size_t)4 * 2 * 32 * 16 * 2);
    unsigned short* xmid  = (unsigned short*)alloc((size_t)BB * HEADS * NPIX * DH * 2);
    unsigned short* fxmid = (unsigned short*)alloc((size_t)BB * HEADS * NPIX * DH * 2);
    unsigned short* sw    = (unsigned short*)alloc((size_t)BB * NPIX * 512 * 2);
    float* tokacc = (float*)alloc((size_t)BB * HEADS * 64 * 64 * 4);
    float* nrm    = (float*)alloc((size_t)BB * HEADS * 64 * 4);
    unsigned short* PB = (unsigned short*)alloc((size_t)BB * 8 * 16 * 512 * 2);

    int nx = BB * NPIX * DIMC;
    k_cvt_x<<<(nx + 255) / 256, 256, 0, stream>>>(x, xb, nx);
    int nwc = 64 * 9 * 4 * 32 * 16;
    k_prep_wconv<<<(nwc + 255) / 256, 256, 0, stream>>>(Wx, Wfx, wcat);
    k_prep_ws<<<16, 256, 0, stream>>>(Ws, wsbuf);
    int nz1 = BB * HEADS * 64 * 64;
    k_zero<<<(nz1 + 255) / 256, 256, 0, stream>>>(tokacc, nz1);
    int nz2 = BB * HEADS * 64;
    k_zero<<<(nz2 + 255) / 256, 256, 0, stream>>>(nrm, nz2);

    k_conv<<<dim3(1024, 8), 256, 0, stream>>>(xb, wcat, bx, bfx, xmid, fxmid);
    k_route<<<dim3(128, 32), 256, 0, stream>>>(xmid, fxmid, wsbuf, bs, temperature, sw, tokacc, nrm);
    k_attn<<<32, 64, 0, stream>>>(tokacc, nrm, Wq, Wk, Wv, Wo, PB);
    k_outproj<<<512, 256, 0, stream>>>(sw, PB, bo, out);
}
